// Block_35416300322903
// MI455X (gfx1250) — matrix operand order repaired and checked by reading, not compiled, not run on hardware
//
#include <hip/hip_runtime.h>
#include <hip/hip_bf16.h>

// ---------------------------------------------------------------------------
// Transformer block (pre-LN, causal MHA, 4x FFN) for gfx1250 / MI455X.
// All GEMMs run on v_wmma_f32_16x16x32_bf16 (f32 accumulate). LN / softmax /
// residuals stay fp32. QKV fused into one N=3072 GEMM; GEMM inner loop is
// software-pipelined (register prefetch of next K=64 stage overlaps WMMA).
// Workspace needs ~216 MB of d_ws.
// B=8, T=1024, C=1024, H=16, D=64, M = B*T = 8192.
// ---------------------------------------------------------------------------

typedef __bf16 bf16_t;
typedef __attribute__((ext_vector_type(8)))  __bf16 bf16x8;
typedef __attribute__((ext_vector_type(16))) __bf16 bf16x16;
typedef __attribute__((ext_vector_type(8)))  float  f32x8;

__device__ __forceinline__ bf16_t f2bf(float f) {
  unsigned int u = __builtin_bit_cast(unsigned int, f);
  unsigned int r = (u + 0x7FFFu + ((u >> 16) & 1u)) >> 16;
  return __builtin_bit_cast(bf16_t, (unsigned short)r);
}

__device__ __forceinline__ f32x8 wmma_bf(bf16x16 a, bf16x16 b, f32x8 c) {
  // (neg_a, A, neg_b, B, c_mod, C, reuse_a, reuse_b)
  return __builtin_amdgcn_wmma_f32_16x16x32_bf16(false, a, false, b,
                                                 (short)0, c, false, false);
}

__device__ __forceinline__ f32x8 zero8() {
  f32x8 z;
#pragma unroll
  for (int i = 0; i < 8; ++i) z[i] = 0.0f;
  return z;
}

// A-matrix fragment (16x32 bf16), tile origin p0, row stride ld (halves).
// lane L<16: m=L, K = {0..7, 16..23}; lane>=16: m=L-16, K = {8..15, 24..31}.
__device__ __forceinline__ bf16x16 load_a_frag(const bf16_t* p0, int ld, int lane) {
  const bf16_t* p = p0 + (size_t)(lane & 15) * ld + ((lane >> 4) << 3);
  bf16x8 lo = *(const bf16x8*)p;
  bf16x8 hi = *(const bf16x8*)(p + 16);
  bf16x16 f;
#pragma unroll
  for (int i = 0; i < 8; ++i) { f[i] = lo[i]; f[i + 8] = hi[i]; }
  return f;
}

// B-matrix fragment (32x16 bf16) from column-major storage: element (k,n) at
// p0[n*ld + k]. lane<16: n=L, K = {0..7, 16..23}; lane>=16: n=L-16, K = {8..15, 24..31}.
__device__ __forceinline__ bf16x16 load_bt_frag(const bf16_t* p0, int ld, int lane) {
  const bf16_t* p = p0 + (size_t)(lane & 15) * ld + ((lane >> 4) << 3);
  bf16x8 lo = *(const bf16x8*)p;
  bf16x8 hi = *(const bf16x8*)(p + 16);
  bf16x16 f;
#pragma unroll
  for (int i = 0; i < 8; ++i) { f[i] = lo[i]; f[i + 8] = hi[i]; }
  return f;
}

__device__ __forceinline__ float red16_max(float v) {
  v = fmaxf(v, __shfl_xor(v, 1));
  v = fmaxf(v, __shfl_xor(v, 2));
  v = fmaxf(v, __shfl_xor(v, 4));
  v = fmaxf(v, __shfl_xor(v, 8));
  return v;
}
__device__ __forceinline__ float red16_sum(float v) {
  v += __shfl_xor(v, 1);
  v += __shfl_xor(v, 2);
  v += __shfl_xor(v, 4);
  v += __shfl_xor(v, 8);
  return v;
}

// ---------------------------------------------------------------------------
// fp32 -> bf16 conversion (contiguous)
// ---------------------------------------------------------------------------
__global__ __launch_bounds__(256) void cvt_bf16_kernel(
    const float* __restrict__ in, bf16_t* __restrict__ out, int n) {
  int i = (blockIdx.x * 256 + threadIdx.x) * 4;
  if (i + 3 < n) {
    float4 v = *(const float4*)(in + i);
    out[i + 0] = f2bf(v.x);
    out[i + 1] = f2bf(v.y);
    out[i + 2] = f2bf(v.z);
    out[i + 3] = f2bf(v.w);
  }
}

// fp32 [rows x cols] -> bf16 strided (for packing Wq|Wk|Wv into [C][3C])
__global__ __launch_bounds__(256) void cvt_bf16_strided_kernel(
    const float* __restrict__ in, bf16_t* __restrict__ out, int cols, int ldo,
    int n) {
  int i = (blockIdx.x * 256 + threadIdx.x) * 4;
  if (i + 3 < n) {
    float4 v = *(const float4*)(in + i);
    int r = i / cols, c = i % cols;
    bf16_t* o = out + (size_t)r * ldo + c;
    o[0] = f2bf(v.x);
    o[1] = f2bf(v.y);
    o[2] = f2bf(v.z);
    o[3] = f2bf(v.w);
  }
}

// ---------------------------------------------------------------------------
// LayerNorm over C=1024, fp32 in -> bf16 out. One 256-thread block per row.
// ---------------------------------------------------------------------------
__global__ __launch_bounds__(256) void layernorm_bf16_kernel(
    const float* __restrict__ x, const float* __restrict__ g,
    const float* __restrict__ bb, bf16_t* __restrict__ out) {
  __shared__ float red[256];
  const int row = blockIdx.x;
  const int tid = threadIdx.x;
  const float* xr = x + (size_t)row * 1024;
  float v[4];
  float s = 0.0f;
#pragma unroll
  for (int i = 0; i < 4; ++i) { v[i] = xr[tid + 256 * i]; s += v[i]; }
  red[tid] = s;
  __syncthreads();
  for (int off = 128; off > 0; off >>= 1) {
    if (tid < off) red[tid] += red[tid + off];
    __syncthreads();
  }
  const float mu = red[0] * (1.0f / 1024.0f);
  __syncthreads();
  float s2 = 0.0f;
#pragma unroll
  for (int i = 0; i < 4; ++i) { float d = v[i] - mu; s2 += d * d; }
  red[tid] = s2;
  __syncthreads();
  for (int off = 128; off > 0; off >>= 1) {
    if (tid < off) red[tid] += red[tid + off];
    __syncthreads();
  }
  const float var = red[0] * (1.0f / 1024.0f);
  const float rs = rsqrtf(var + 1e-5f);
  bf16_t* orow = out + (size_t)row * 1024;
#pragma unroll
  for (int i = 0; i < 4; ++i) {
    int c = tid + 256 * i;
    orow[c] = f2bf((v[i] - mu) * rs * g[c] + bb[c]);
  }
}

// ---------------------------------------------------------------------------
// WMMA GEMM: out = A[MxK] @ B[KxN] (+bias) (+relu) (+residual).
// 128x128 block tile, 256 threads = 8 waves, each wave 64x32 (4x2 WMMA tiles).
// K-step 64, software-pipelined: next stage's global tiles prefetched into
// registers while the current stage's 16 WMMAs run from LDS.
// RES => fp32 output (acc + bias + residual), else bf16 output.
// K must be a multiple of 64; M,N multiples of 128.
// ---------------------------------------------------------------------------
template <bool BIAS, bool RELU, bool RES>
__global__ __launch_bounds__(256) void gemm_bf16_kernel(
    const bf16_t* __restrict__ A, const bf16_t* __restrict__ B,
    const float* __restrict__ bias, const float* __restrict__ res,
    bf16_t* __restrict__ outb, float* __restrict__ outf,
    int M, int N, int K) {
  __shared__ bf16_t As[128 * 72];   // row-major, padded stride 72 halves
  __shared__ bf16_t Bs[128 * 72];   // B^T: Bs[n*72 + k]
  const int tid  = threadIdx.x;
  const int lane = tid & 31;
  const int w    = tid >> 5;
  const int m0   = blockIdx.y * 128;
  const int n0   = blockIdx.x * 128;
  const int wm   = (w & 1) * 64;
  const int wn   = (w >> 1) * 32;

  f32x8 acc[4][2];
#pragma unroll
  for (int i = 0; i < 4; ++i)
#pragma unroll
    for (int j = 0; j < 2; ++j) acc[i][j] = zero8();

  const int arow = tid >> 1, acol = (tid & 1) << 5;   // 128 rows x 64 halves
  const int brow = tid >> 2, bcol = (tid & 3) << 5;   // 64 rows x 128 halves

  bf16x8 areg[4], breg[4];
  {
    const bf16_t* ap = A + (size_t)(m0 + arow) * K + acol;
    const bf16_t* bp = B + (size_t)brow * N + n0 + bcol;
#pragma unroll
    for (int i = 0; i < 4; ++i) {
      areg[i] = *(const bf16x8*)(ap + 8 * i);
      breg[i] = *(const bf16x8*)(bp + 8 * i);
    }
  }

  for (int kt = 0; kt < K; kt += 64) {
    __syncthreads();
#pragma unroll
    for (int i = 0; i < 4; ++i)
      *(bf16x8*)&As[arow * 72 + acol + 8 * i] = areg[i];
#pragma unroll
    for (int i = 0; i < 4; ++i)
#pragma unroll
      for (int j = 0; j < 8; ++j)
        Bs[(bcol + 8 * i + j) * 72 + brow] = breg[i][j];
    __syncthreads();

    if (kt + 64 < K) {  // prefetch next stage; overlaps the WMMAs below
      const bf16_t* ap = A + (size_t)(m0 + arow) * K + kt + 64 + acol;
      const bf16_t* bp = B + (size_t)(kt + 64 + brow) * N + n0 + bcol;
#pragma unroll
      for (int i = 0; i < 4; ++i) {
        areg[i] = *(const bf16x8*)(ap + 8 * i);
        breg[i] = *(const bf16x8*)(bp + 8 * i);
      }
    }

#pragma unroll
    for (int kc = 0; kc < 2; ++kc) {
      bf16x16 afr[4], bfr[2];
#pragma unroll
      for (int i = 0; i < 4; ++i)
        afr[i] = load_a_frag(&As[(wm + 16 * i) * 72 + kc * 32], 72, lane);
#pragma unroll
      for (int j = 0; j < 2; ++j)
        bfr[j] = load_bt_frag(&Bs[(wn + 16 * j) * 72 + kc * 32], 72, lane);
#pragma unroll
      for (int i = 0; i < 4; ++i)
#pragma unroll
        for (int j = 0; j < 2; ++j)
          acc[i][j] = wmma_bf(afr[i], bfr[j], acc[i][j]);
    }
  }

  const int hi = (lane >> 4) << 3;
  const int nn = lane & 15;
#pragma unroll
  for (int i = 0; i < 4; ++i) {
#pragma unroll
    for (int j = 0; j < 2; ++j) {
#pragma unroll
      for (int r = 0; r < 8; ++r) {
        const int gm = m0 + wm + 16 * i + r + hi;
        const int gn = n0 + wn + 16 * j + nn;
        float v = acc[i][j][r];
        if (BIAS) v += bias[gn];
        if (RELU) v = fmaxf(v, 0.0f);
        const size_t idx = (size_t)gm * N + gn;
        if (RES) outf[idx] = v + res[idx];
        else     outb[idx] = f2bf(v);
      }
    }
  }
}

// ---------------------------------------------------------------------------
// Flash attention, causal. Grid (B*H, T/64), block 128 (4 waves).
// QKV fused buffer: [B*T][3072], Q at col 0, K at 1024, V at 2048 (+h*64).
// Wave w owns 16 query rows; loop over 32-key strips up to the block diagonal.
// ---------------------------------------------------------------------------
#define QKV_LD 3072

__global__ __launch_bounds__(128) void attention_kernel(
    const bf16_t* __restrict__ QKV, bf16_t* __restrict__ Og) {
  __shared__ bf16_t Vt[64 * 40];      // Vt[d*40 + k] : V strip transposed
  __shared__ bf16_t Pb[4][16 * 40];   // per-wave P (16x32, stride 40)

  const int lane = threadIdx.x & 31;
  const int w    = threadIdx.x >> 5;
  const int b    = blockIdx.x >> 4;
  const int h    = blockIdx.x & 15;
  const int q0   = blockIdx.y * 64;
  const int qw   = q0 + w * 16;
  const size_t rowbase = (size_t)b * 1024;           // row index base (b*T)
  const bf16_t* Qbase = QKV + (rowbase + qw) * QKV_LD + h * 64;

  bf16x16 qf[2];
  qf[0] = load_a_frag(Qbase, QKV_LD, lane);
  qf[1] = load_a_frag(Qbase + 32, QKV_LD, lane);

  f32x8 O[4];
#pragma unroll
  for (int i = 0; i < 4; ++i) O[i] = zero8();
  float mrun[8], lrun[8];
#pragma unroll
  for (int r = 0; r < 8; ++r) { mrun[r] = -1e30f; lrun[r] = 0.0f; }

  const int nstrips = (q0 + 64) >> 5;
  const int nn   = lane & 15;
  const int hi8  = (lane >> 4) << 3;
  bf16_t* Pw = &Pb[w][0];

  for (int t = 0; t < nstrips; ++t) {
    const int s2 = t << 5;
    __syncthreads();                       // Vt reuse from previous strip
    {
      const int k  = threadIdx.x >> 2;           // 0..31
      const int d0 = (threadIdx.x & 3) << 4;     // 0,16,32,48
      const bf16_t* src =
          QKV + (rowbase + s2 + k) * QKV_LD + 2048 + h * 64 + d0;
      bf16x8 v0 = *(const bf16x8*)src;
      bf16x8 v1 = *(const bf16x8*)(src + 8);
#pragma unroll
      for (int j = 0; j < 8; ++j) {
        Vt[(d0 + j) * 40 + k]     = v0[j];
        Vt[(d0 + 8 + j) * 40 + k] = v1[j];
      }
    }
    __syncthreads();

    // S = Q @ K^T for two 16-key sub-tiles (K dim 64 = 2 WMMA chunks)
    const bf16_t* Kb = QKV + (rowbase + s2) * QKV_LD + 1024 + h * 64;
    f32x8 S0 = zero8(), S1 = zero8();
#pragma unroll
    for (int kc = 0; kc < 2; ++kc) {
      bf16x16 kf0 = load_bt_frag(Kb + kc * 32, QKV_LD, lane);
      bf16x16 kf1 = load_bt_frag(Kb + (size_t)16 * QKV_LD + kc * 32, QKV_LD, lane);
      S0 = wmma_bf(qf[kc], kf0, S0);
      S1 = wmma_bf(qf[kc], kf1, S1);
    }

    // scale, causal mask, online softmax; write P to per-wave LDS
#pragma unroll
    for (int r = 0; r < 8; ++r) {
      const int m  = r + hi8;
      const int qi = qw + m;
      float s0 = S0[r] * 0.125f;
      float s1 = S1[r] * 0.125f;
      if (s2 + nn      > qi) s0 = -1e30f;
      if (s2 + 16 + nn > qi) s1 = -1e30f;
      const float mx   = red16_max(fmaxf(s0, s1));
      const float mnew = fmaxf(mrun[r], mx);
      const float alpha = __expf(mrun[r] - mnew);
      const float p0 = __expf(s0 - mnew);
      const float p1 = __expf(s1 - mnew);
      const float rs = red16_sum(p0 + p1);
      lrun[r] = lrun[r] * alpha + rs;
      mrun[r] = mnew;
#pragma unroll
      for (int vb = 0; vb < 4; ++vb) O[vb][r] *= alpha;
      Pw[m * 40 + nn]      = f2bf(p0);
      Pw[m * 40 + 16 + nn] = f2bf(p1);
    }
    __syncthreads();                       // P cross-lane visibility

    // O += P(16x32) @ V(32x64)
    bf16x16 pf = load_a_frag(Pw, 40, lane);
#pragma unroll
    for (int vb = 0; vb < 4; ++vb) {
      bf16x16 vf = load_bt_frag(&Vt[(vb * 16) * 40], 40, lane);
      O[vb] = wmma_bf(pf, vf, O[vb]);
    }
  }

  // normalize + write attn (bf16) in [B*T, H*D] layout (stride 1024)
#pragma unroll
  for (int r = 0; r < 8; ++r) {
    const int m = r + hi8;
    const float inv = 1.0f / lrun[r];
    const size_t row = (rowbase + qw + m) * 1024 + h * 64;
#pragma unroll
    for (int vb = 0; vb < 4; ++vb)
      Og[row + vb * 16 + nn] = f2bf(O[vb][r] * inv);
  }
}

// ---------------------------------------------------------------------------
// Host launch
// ---------------------------------------------------------------------------
extern "C" void kernel_launch(void* const* d_in, const int* in_sizes, int n_in,
                              void* d_out, int out_size, void* d_ws, size_t ws_size,
                              hipStream_t stream) {
  (void)in_sizes; (void)n_in; (void)out_size; (void)ws_size;
  const float* x     = (const float*)d_in[0];
  const float* Wq    = (const float*)d_in[1];
  const float* Wk    = (const float*)d_in[2];
  const float* Wv    = (const float*)d_in[3];
  const float* Wp    = (const float*)d_in[4];
  const float* bp    = (const float*)d_in[5];
  const float* ln1_g = (const float*)d_in[6];
  const float* ln1_b = (const float*)d_in[7];
  const float* ln2_g = (const float*)d_in[8];
  const float* ln2_b = (const float*)d_in[9];
  const float* W1    = (const float*)d_in[10];
  const float* b1    = (const float*)d_in[11];
  const float* W2    = (const float*)d_in[12];
  const float* b2    = (const float*)d_in[13];

  const int M = 8192, C = 1024, F = 4096, C3 = 3072;
  const size_t MB = 1ull << 20;
  char* ws = (char*)d_ws;
  bf16_t* WqkvB = (bf16_t*)(ws + 0 * MB);    // [C][3C] bf16, 6 MB
  bf16_t* WpB   = (bf16_t*)(ws + 6 * MB);    // 2 MB
  bf16_t* W1B   = (bf16_t*)(ws + 8 * MB);    // 8 MB
  bf16_t* W2B   = (bf16_t*)(ws + 16 * MB);   // 8 MB
  bf16_t* XN    = (bf16_t*)(ws + 24 * MB);   // 16 MB
  bf16_t* QKV   = (bf16_t*)(ws + 40 * MB);   // [M][3C] bf16, 48 MB
  bf16_t* AT    = (bf16_t*)(ws + 88 * MB);   // 16 MB
  float*  X1    = (float*)(ws + 104 * MB);   // 32 MB
  bf16_t* XN2   = (bf16_t*)(ws + 136 * MB);  // 16 MB
  bf16_t* Hh    = (bf16_t*)(ws + 152 * MB);  // 64 MB

  // 1) weights fp32 -> bf16 (Wq|Wk|Wv packed into one [C][3C] matrix)
  cvt_bf16_strided_kernel<<<C * C / 1024, 256, 0, stream>>>(Wq, WqkvB + 0,
                                                            C, C3, C * C);
  cvt_bf16_strided_kernel<<<C * C / 1024, 256, 0, stream>>>(Wk, WqkvB + C,
                                                            C, C3, C * C);
  cvt_bf16_strided_kernel<<<C * C / 1024, 256, 0, stream>>>(Wv, WqkvB + 2 * C,
                                                            C, C3, C * C);
  cvt_bf16_kernel<<<C * C / 1024, 256, 0, stream>>>(Wp, WpB, C * C);
  cvt_bf16_kernel<<<C * F / 1024, 256, 0, stream>>>(W1, W1B, C * F);
  cvt_bf16_kernel<<<C * F / 1024, 256, 0, stream>>>(W2, W2B, F * C);

  // 2) LN1
  layernorm_bf16_kernel<<<M, 256, 0, stream>>>(x, ln1_g, ln1_b, XN);

  // 3) fused QKV projection: [M][C] @ [C][3C] -> [M][3C]
  dim3 gQKV(C3 / 128, M / 128);
  gemm_bf16_kernel<false, false, false><<<gQKV, 256, 0, stream>>>(
      XN, WqkvB, nullptr, nullptr, QKV, nullptr, M, C3, C);

  // 4) causal flash attention
  attention_kernel<<<dim3(8 * 16, 1024 / 64), 128, 0, stream>>>(QKV, AT);

  // 5) output projection + bias + residual -> X1 (fp32)
  dim3 gC(C / 128, M / 128);
  gemm_bf16_kernel<true, false, true><<<gC, 256, 0, stream>>>(
      AT, WpB, bp, x, nullptr, X1, M, C, C);

  // 6) LN2
  layernorm_bf16_kernel<<<M, 256, 0, stream>>>(X1, ln2_g, ln2_b, XN2);

  // 7) FFN up + ReLU
  dim3 gF(F / 128, M / 128);
  gemm_bf16_kernel<true, true, false><<<gF, 256, 0, stream>>>(
      XN2, W1B, b1, nullptr, Hh, nullptr, M, F, C);

  // 8) FFN down + bias + residual -> out (fp32)
  gemm_bf16_kernel<true, false, true><<<gC, 256, 0, stream>>>(
      Hh, W2B, b2, X1, nullptr, (float*)d_out, M, C, F);
}
